// SB_nl_34754875359795
// MI455X (gfx1250) — matrix operand order repaired and checked by reading, not compiled, not run on hardware
//
#include <hip/hip_runtime.h>
#include <hip/hip_bf16.h>

// ---------------------------------------------------------------------------
// CDNA5 (gfx1250) non-local attention block, bf16 WMMA pipeline.
// Big GEMMs: 16x64 per-wave strips (1x4 N-blocking) with explicit software
// pipelining so next-iteration fragment loads overlap current WMMAs.
// ---------------------------------------------------------------------------

typedef __attribute__((ext_vector_type(16))) __bf16 v16bf;
typedef __attribute__((ext_vector_type(8)))  __bf16 v8bf;
typedef __attribute__((ext_vector_type(8)))  float  v8f;

#define DEV static __device__ __forceinline__

namespace nl {
constexpr int B    = 32;
constexpr int C    = 1024;
constexpr int CI   = 512;
constexpr int T    = 4;
constexpr int H    = 14;
constexpr int W    = 14;
constexpr int N    = T * H * W;     // 784 = 49 * 16
constexpr int NS   = 196;           // pooled tokens
constexpr int NSP  = 208;           // NS padded to 16  (13 tiles)
constexpr int NSP2 = 224;           // NS padded to 32  (K for y-GEMM)
}

DEV int lane_id() { return (int)(threadIdx.x & 31u); }

// A fragment: 16x32 bf16, row-major source, lda in elements.
// ISA 7.12.2: lane<16 -> row=lane, K = {k0..k0+7, k0+16..k0+23}
//             lane>=16 -> row=lane-16, K shifted by +8.
DEV v16bf load_a_16x32(const __bf16* __restrict__ A, int lda, int k0) {
  const int lane = lane_id();
  const int row  = lane & 15;
  const int kb   = (lane < 16) ? 0 : 8;
  const __bf16* p = A + (size_t)row * lda + k0 + kb;
  v8bf lo = *(const v8bf*)(p);
  v8bf hi = *(const v8bf*)(p + 16);
  v16bf r;
#pragma unroll
  for (int i = 0; i < 8; ++i) { r[i] = lo[i]; r[8 + i] = hi[i]; }
  return r;
}

// B fragment: 32x16 bf16 where memory holds B^T row-major [Ncols][K].
// ISA: lane<16 -> col=lane, K = {k0..k0+7, k0+16..k0+23}; lane>=16 -> col=lane-16, K shifted by +8.
DEV v16bf load_bT_32x16(const __bf16* __restrict__ BT, int ldb, int k0) {
  const int lane = lane_id();
  const int col  = lane & 15;
  const int kb   = (lane < 16) ? 0 : 8;
  const __bf16* p = BT + (size_t)col * ldb + k0 + kb;
  v8bf lo = *(const v8bf*)(p);
  v8bf hi = *(const v8bf*)(p + 16);
  v16bf r;
#pragma unroll
  for (int i = 0; i < 8; ++i) { r[i] = lo[i]; r[8 + i] = hi[i]; }
  return r;
}

DEV v8f wmma_bf16(const v16bf& a, const v16bf& b, const v8f& c) {
  return __builtin_amdgcn_wmma_f32_16x16x32_bf16(
      false, a, false, b, (short)0, c, false, false);
}

// Single 16x16 tile, software-pipelined: loads for step k+1 issue before the
// WMMA of step k so the compiler can hold partial loadcnt waits.
DEV v8f wmma_gemm_tile(const __bf16* __restrict__ A, int lda,
                       const __bf16* __restrict__ BT, int ldb, int K) {
  v8f acc = {0.f, 0.f, 0.f, 0.f, 0.f, 0.f, 0.f, 0.f};
  v16bf a = load_a_16x32(A, lda, 0);
  v16bf b = load_bT_32x16(BT, ldb, 0);
  for (int k0 = 32; k0 < K; k0 += 32) {
    v16bf an = load_a_16x32(A, lda, k0);
    v16bf bn = load_bT_32x16(BT, ldb, k0);
    acc = wmma_bf16(a, b, acc);
    a = an; b = bn;
  }
  return wmma_bf16(a, b, acc);
}

// 16x64 strip: 4 accumulators, A fragment reused across 4 v_wmma per K-step.
// Software-pipelined: next-step a/b fragments are loaded before current WMMAs.
DEV void wmma_gemm_strip4(const __bf16* __restrict__ A, int lda,
                          const __bf16* __restrict__ BT, int ldb, int K,
                          v8f acc[4]) {
#pragma unroll
  for (int j = 0; j < 4; ++j) acc[j] = (v8f){0.f,0.f,0.f,0.f,0.f,0.f,0.f,0.f};
  v16bf a  = load_a_16x32(A, lda, 0);
  v16bf b0 = load_bT_32x16(BT + (size_t)0 * 16 * ldb, ldb, 0);
  v16bf b1 = load_bT_32x16(BT + (size_t)1 * 16 * ldb, ldb, 0);
  v16bf b2 = load_bT_32x16(BT + (size_t)2 * 16 * ldb, ldb, 0);
  v16bf b3 = load_bT_32x16(BT + (size_t)3 * 16 * ldb, ldb, 0);
  for (int k0 = 32; k0 < K; k0 += 32) {
    v16bf an  = load_a_16x32(A, lda, k0);
    v16bf bn0 = load_bT_32x16(BT + (size_t)0 * 16 * ldb, ldb, k0);
    v16bf bn1 = load_bT_32x16(BT + (size_t)1 * 16 * ldb, ldb, k0);
    v16bf bn2 = load_bT_32x16(BT + (size_t)2 * 16 * ldb, ldb, k0);
    v16bf bn3 = load_bT_32x16(BT + (size_t)3 * 16 * ldb, ldb, k0);
    acc[0] = wmma_bf16(a, b0, acc[0]);
    acc[1] = wmma_bf16(a, b1, acc[1]);
    acc[2] = wmma_bf16(a, b2, acc[2]);
    acc[3] = wmma_bf16(a, b3, acc[3]);
    a = an; b0 = bn0; b1 = bn1; b2 = bn2; b3 = bn3;
  }
  acc[0] = wmma_bf16(a, b0, acc[0]);
  acc[1] = wmma_bf16(a, b1, acc[1]);
  acc[2] = wmma_bf16(a, b2, acc[2]);
  acc[3] = wmma_bf16(a, b3, acc[3]);
}

// ---------------------------------------------------------------------------
// Kernel 0: f32 -> bf16 convert (weights)
// ---------------------------------------------------------------------------
__global__ void __launch_bounds__(256)
nl_cvt_bf16_kernel(const float* __restrict__ src, __bf16* __restrict__ dst, int n) {
  int i = blockIdx.x * 256 + threadIdx.x;
  if (i < n) dst[i] = (__bf16)src[i];
}

// ---------------------------------------------------------------------------
// Kernel 1: x [B,C,N] f32 -> xt [B,N,C] bf16 (coalesced reads)
// ---------------------------------------------------------------------------
__global__ void __launch_bounds__(256)
nl_xpose_kernel(const float* __restrict__ x, __bf16* __restrict__ xt) {
  using namespace nl;
  size_t i = (size_t)blockIdx.x * 256 + threadIdx.x;   // over B*C*N
  int n = (int)(i % N);
  size_t bc = i / N;
  int c = (int)(bc % C);
  int b = (int)(bc / C);
  xt[((size_t)b * N + n) * C + c] = (__bf16)x[i];
}

// ---------------------------------------------------------------------------
// Kernel 2: projections theta/phi/g = xt @ W^T + bias   (bf16 out)
// grid = (49, 3, 32), block = 256: 8 waves, each a 16x64 strip.
// ---------------------------------------------------------------------------
__global__ void __launch_bounds__(256)
nl_proj_gemm_kernel(const __bf16* __restrict__ xt,
                    const __bf16* __restrict__ wt, const __bf16* __restrict__ wp,
                    const __bf16* __restrict__ wg,
                    const float* __restrict__ bt, const float* __restrict__ bp,
                    const float* __restrict__ bg,
                    __bf16* __restrict__ ot, __bf16* __restrict__ op,
                    __bf16* __restrict__ og) {
  using namespace nl;
  const int b    = blockIdx.z;
  const int proj = blockIdx.y;
  const int wave = threadIdx.x >> 5;
  const int u  = blockIdx.x * 8 + wave;        // 0..391, exact
  const int mt = u >> 3;                       // 49 m-tiles
  const int nb = u & 7;                        // 8 n-blocks of 64

  const __bf16* Wm; const float* bias; __bf16* out;
  if (proj == 0)      { Wm = wt; bias = bt; out = ot; }
  else if (proj == 1) { Wm = wp; bias = bp; out = op; }
  else                { Wm = wg; bias = bg; out = og; }

  const __bf16* A  = xt + (size_t)b * N * C + (size_t)mt * 16 * C;
  const __bf16* BT = Wm + (size_t)nb * 64 * C;
  v8f acc[4];
  wmma_gemm_strip4(A, C, BT, C, C, acc);

  const int lane = lane_id();
  const int col  = lane & 15;
  const int rb   = (lane < 16) ? 0 : 8;
  __bf16* D = out + (size_t)b * N * CI + (size_t)mt * 16 * CI + nb * 64;
#pragma unroll
  for (int j = 0; j < 4; ++j) {
    const float bv = bias[nb * 64 + j * 16 + col];
#pragma unroll
    for (int e = 0; e < 8; ++e)
      D[(size_t)(rb + e) * CI + j * 16 + col] = (__bf16)(acc[j][e] + bv);
  }
}

// ---------------------------------------------------------------------------
// Kernel 3: 2x2 spatial max-pool on phi/g tokens.
// phi_p: [B][NSP][CI] row-major (B^T layout for f-GEMM), pad rows zeroed.
// g_pT : [B][CI][NSP2]          (B^T layout for y-GEMM), pad cols zeroed.
// ---------------------------------------------------------------------------
DEV float pool_max4(const __bf16* __restrict__ src, int n0, int ci) {
  using namespace nl;
  float a = (float)src[(size_t)n0 * CI + ci];
  float b = (float)src[(size_t)(n0 + 1) * CI + ci];
  float c = (float)src[(size_t)(n0 + W) * CI + ci];
  float d = (float)src[(size_t)(n0 + W + 1) * CI + ci];
  return fmaxf(fmaxf(a, b), fmaxf(c, d));
}

__global__ void __launch_bounds__(256)
nl_pool_kernel(const __bf16* __restrict__ phif, const __bf16* __restrict__ gf,
               __bf16* __restrict__ phip, __bf16* __restrict__ gpt) {
  using namespace nl;
  size_t i = (size_t)blockIdx.x * 256 + threadIdx.x;   // over B*CI*NSP2
  int m  = (int)(i % NSP2);
  size_t r = i / NSP2;
  int ci = (int)(r % CI);
  int b  = (int)(r / CI);

  float pv = 0.f, gv = 0.f;
  if (m < NS) {
    int t  = m / 49;
    int rr = m % 49;
    int hh = rr / 7, ww = rr % 7;
    int n0 = t * (H * W) + hh * 2 * W + ww * 2;
    const __bf16* pf = phif + (size_t)b * N * CI;
    const __bf16* gg = gf   + (size_t)b * N * CI;
    pv = pool_max4(pf, n0, ci);
    gv = pool_max4(gg, n0, ci);
  }
  gpt[((size_t)b * CI + ci) * NSP2 + m] = (__bf16)gv;
  if (m < NSP)
    phip[((size_t)b * NSP + m) * CI + ci] = (__bf16)pv;
}

// ---------------------------------------------------------------------------
// Kernel 4: f[784, NSP] = theta @ phi_p^T   (f32 out)
// grid = (80, 32); 637 tiles per batch (guarded); 1x1 tiles (N=13 tiles).
// ---------------------------------------------------------------------------
__global__ void __launch_bounds__(256)
nl_f_gemm_kernel(const __bf16* __restrict__ theta,
                 const __bf16* __restrict__ phip, float* __restrict__ f) {
  using namespace nl;
  const int b    = blockIdx.y;
  const int wave = threadIdx.x >> 5;
  const int tile = blockIdx.x * 8 + wave;
  if (tile >= 49 * 13) return;                 // wave-uniform
  const int mt = tile / 13, nt = tile % 13;

  const __bf16* A  = theta + (size_t)b * N * CI + (size_t)mt * 16 * CI;
  const __bf16* BT = phip  + (size_t)b * NSP * CI + (size_t)nt * 16 * CI;
  v8f acc = wmma_gemm_tile(A, CI, BT, CI, CI);

  const int lane = lane_id();
  const int col  = lane & 15;
  const int rb   = (lane < 16) ? 0 : 8;
  float* D = f + (size_t)b * N * NSP + (size_t)mt * 16 * NSP + nt * 16;
#pragma unroll
  for (int e = 0; e < 8; ++e) D[(size_t)(rb + e) * NSP + col] = acc[e];
}

// ---------------------------------------------------------------------------
// Kernel 5: row softmax over 196 valid cols -> attn bf16 [*, NSP2], pads = 0.
// One wave per row; 8 rows per block; grid = 25088/8 = 3136.
// ---------------------------------------------------------------------------
__global__ void __launch_bounds__(256)
nl_softmax_kernel(const float* __restrict__ f, __bf16* __restrict__ attn) {
  using namespace nl;
  const int wave = threadIdx.x >> 5;
  const int lane = lane_id();
  const int row  = blockIdx.x * 8 + wave;      // over B*N rows
  const float* fr = f + (size_t)row * NSP;
  __bf16* ar = attn + (size_t)row * NSP2;

  float m = -INFINITY;
  for (int i = lane; i < NS; i += 32) m = fmaxf(m, fr[i]);
#pragma unroll
  for (int off = 16; off >= 1; off >>= 1) m = fmaxf(m, __shfl_xor(m, off, 32));

  float e[7];
  float s = 0.f;
  int cnt = 0;
  for (int i = lane; i < NS; i += 32) { float t = __expf(fr[i] - m); e[cnt++] = t; s += t; }
#pragma unroll
  for (int off = 16; off >= 1; off >>= 1) s += __shfl_xor(s, off, 32);
  const float inv = 1.f / s;

  cnt = 0;
  for (int i = lane; i < NS; i += 32) ar[i] = (__bf16)(e[cnt++] * inv);
  for (int i = NS + lane; i < NSP2; i += 32) ar[i] = (__bf16)0.f;
}

// ---------------------------------------------------------------------------
// Kernel 6: y[784, CI] = attn[784, NSP2] @ g  (B^T = g_pT [CI][NSP2]); bf16 out
// grid = (49, 32): 392 wave-units = 49 m-tiles x 8 n-blocks of 64.
// ---------------------------------------------------------------------------
__global__ void __launch_bounds__(256)
nl_y_gemm_kernel(const __bf16* __restrict__ attn,
                 const __bf16* __restrict__ gpt, __bf16* __restrict__ y) {
  using namespace nl;
  const int b    = blockIdx.y;
  const int wave = threadIdx.x >> 5;
  const int u  = blockIdx.x * 8 + wave;        // 0..391, exact
  const int mt = u >> 3;
  const int nb = u & 7;

  const __bf16* A  = attn + (size_t)b * N * NSP2 + (size_t)mt * 16 * NSP2;
  const __bf16* BT = gpt  + (size_t)b * CI * NSP2 + (size_t)nb * 64 * NSP2;
  v8f acc[4];
  wmma_gemm_strip4(A, NSP2, BT, NSP2, NSP2, acc);

  const int lane = lane_id();
  const int col  = lane & 15;
  const int rb   = (lane < 16) ? 0 : 8;
  __bf16* D = y + (size_t)b * N * CI + (size_t)mt * 16 * CI + nb * 64;
#pragma unroll
  for (int j = 0; j < 4; ++j)
#pragma unroll
    for (int e = 0; e < 8; ++e)
      D[(size_t)(rb + e) * CI + j * 16 + col] = (__bf16)acc[j][e];
}

// ---------------------------------------------------------------------------
// Kernel 7: zero pooled accumulator
// ---------------------------------------------------------------------------
__global__ void __launch_bounds__(256)
nl_zero_kernel(float* __restrict__ p, int n) {
  int i = blockIdx.x * 256 + threadIdx.x;
  if (i < n) p[i] = 0.f;
}

// ---------------------------------------------------------------------------
// Kernel 8: wy = y @ w_out^T + b_out, BN(eval), + residual x, column-sum into
// pooled[b][c] via atomics. grid = (98, 32): 784 wave-units = 49 x 16 blocks.
// ---------------------------------------------------------------------------
__global__ void __launch_bounds__(256)
nl_out_gemm_kernel(const __bf16* __restrict__ y, const __bf16* __restrict__ wo,
                   const float* __restrict__ bo,
                   const float* __restrict__ gamma, const float* __restrict__ beta,
                   const float* __restrict__ mean,  const float* __restrict__ var,
                   const float* __restrict__ x, float* __restrict__ pooled) {
  using namespace nl;
  const int b    = blockIdx.y;
  const int wave = threadIdx.x >> 5;
  const int u  = blockIdx.x * 8 + wave;        // 0..783, exact
  const int mt = u >> 4;                       // 49 m-tiles
  const int nb = u & 15;                       // 16 n-blocks of 64 (C=1024)

  const __bf16* A  = y  + (size_t)b * N * CI + (size_t)mt * 16 * CI;
  const __bf16* BT = wo + (size_t)nb * 64 * CI;
  v8f acc[4];
  wmma_gemm_strip4(A, CI, BT, CI, CI, acc);

  const int lane = lane_id();
  const int col  = lane & 15;
  const int rb   = (lane < 16) ? 0 : 8;

#pragma unroll
  for (int j = 0; j < 4; ++j) {
    const int c = nb * 64 + j * 16 + col;
    const float scale = gamma[c] * rsqrtf(var[c] + 1e-5f);
    const float shift = beta[c];
    const float mu    = mean[c];
    const float bias  = bo[c];
    const float* xrow = x + ((size_t)b * C + c) * N + mt * 16 + rb;
    float sum = 0.f;
#pragma unroll
    for (int e = 0; e < 8; ++e) {
      float v = (acc[j][e] + bias - mu) * scale + shift + xrow[e];
      sum += v;
    }
    atomicAdd(&pooled[(size_t)b * C + c], sum);
  }
}

// ---------------------------------------------------------------------------
// Kernel 9: out[b][j] = mean_n(z)[c] . fc_w[j][c] + fc_b[j]
// ---------------------------------------------------------------------------
__global__ void __launch_bounds__(64)
nl_final_kernel(const float* __restrict__ pooled, const float* __restrict__ fcw,
                const float* __restrict__ fcb, float* __restrict__ out) {
  using namespace nl;
  int t = threadIdx.x;
  if (t >= B * 2) return;
  int b = t >> 1, j = t & 1;
  float acc = 0.f;
  for (int c = 0; c < C; ++c)
    acc += pooled[(size_t)b * C + c] * fcw[(size_t)j * C + c];
  out[t] = acc * (1.0f / (float)N) + fcb[j];
}

// ---------------------------------------------------------------------------
// Host launcher
// ---------------------------------------------------------------------------
extern "C" void kernel_launch(void* const* d_in, const int* in_sizes, int n_in,
                              void* d_out, int out_size, void* d_ws, size_t ws_size,
                              hipStream_t stream) {
  using namespace nl;
  (void)in_sizes; (void)n_in; (void)out_size; (void)ws_size;

  const float* x        = (const float*)d_in[0];
  const float* w_theta  = (const float*)d_in[1];
  const float* b_theta  = (const float*)d_in[2];
  const float* w_phi    = (const float*)d_in[3];
  const float* b_phi    = (const float*)d_in[4];
  const float* w_g      = (const float*)d_in[5];
  const float* b_g      = (const float*)d_in[6];
  const float* w_out    = (const float*)d_in[7];
  const float* b_out    = (const float*)d_in[8];
  const float* bn_gamma = (const float*)d_in[9];
  const float* bn_beta  = (const float*)d_in[10];
  const float* bn_mean  = (const float*)d_in[11];
  const float* bn_var   = (const float*)d_in[12];
  const float* fc_w     = (const float*)d_in[13];
  const float* fc_b     = (const float*)d_in[14];
  float* out = (float*)d_out;

  char* ws = (char*)d_ws;
  // workspace layout (byte offsets); buffers reused once their producers die:
  const size_t o_xt = 0;                                         // xt bf16 [B,N,C]; later y bf16 [B,N,CI]
  const size_t o_w  = o_xt + (size_t)B * N * C * 2;              // 4 bf16 weight mats
  const size_t o_th = o_w  + (size_t)4 * CI * C * 2;             // theta bf16
  const size_t o_pf = o_th + (size_t)B * N * CI * 2;             // phi_full bf16; later f f32 [B,N,NSP]
  const size_t o_gf = o_pf + (size_t)B * N * CI * 2;             // g_full bf16; later attn bf16 [B,N,NSP2]
  const size_t o_pp = o_gf + (size_t)B * N * CI * 2;             // phi_p bf16 [B,NSP,CI]
  const size_t o_gt = o_pp + (size_t)B * NSP * CI * 2;           // g_pT bf16 [B,CI,NSP2]
  const size_t o_po = o_gt + (size_t)B * CI * NSP2 * 2;          // pooled f32 [B,C]

  __bf16* xt_bf = (__bf16*)(ws + o_xt);
  __bf16* wt_bf = (__bf16*)(ws + o_w);
  __bf16* wp_bf = wt_bf + (size_t)CI * C;
  __bf16* wg_bf = wp_bf + (size_t)CI * C;
  __bf16* wo_bf = wg_bf + (size_t)CI * C;
  __bf16* th_bf = (__bf16*)(ws + o_th);
  __bf16* pf_bf = (__bf16*)(ws + o_pf);
  __bf16* gf_bf = (__bf16*)(ws + o_gf);
  __bf16* pp_bf = (__bf16*)(ws + o_pp);
  __bf16* gt_bf = (__bf16*)(ws + o_gt);
  float*  f_f32 = (float*)(ws + o_pf);    // overlays phi_full (dead after pool)
  __bf16* at_bf = (__bf16*)(ws + o_gf);   // overlays g_full   (dead after pool)
  __bf16* y_bf  = (__bf16*)(ws + o_xt);   // overlays xt       (dead after proj)
  float*  pooled = (float*)(ws + o_po);

  // 0) weights -> bf16
  {
    const int nW = CI * C;                       // 524288
    dim3 g(nW / 256);
    nl_cvt_bf16_kernel<<<g, 256, 0, stream>>>(w_theta, wt_bf, nW);
    nl_cvt_bf16_kernel<<<g, 256, 0, stream>>>(w_phi,   wp_bf, nW);
    nl_cvt_bf16_kernel<<<g, 256, 0, stream>>>(w_g,     wg_bf, nW);
    nl_cvt_bf16_kernel<<<g, 256, 0, stream>>>(w_out,   wo_bf, nW);
  }
  // 1) x -> token-major bf16
  nl_xpose_kernel<<<dim3((B * (size_t)C * N) / 256), 256, 0, stream>>>(x, xt_bf);
  // 2) projections (WMMA, 16x64 strips)
  nl_proj_gemm_kernel<<<dim3(49, 3, B), 256, 0, stream>>>(
      xt_bf, wt_bf, wp_bf, wg_bf, b_theta, b_phi, b_g, th_bf, pf_bf, gf_bf);
  // 3) max-pool phi/g
  nl_pool_kernel<<<dim3((B * CI * NSP2) / 256), 256, 0, stream>>>(
      pf_bf, gf_bf, pp_bf, gt_bf);
  // 4) f = theta @ phi_p^T (WMMA)
  nl_f_gemm_kernel<<<dim3(80, B), 256, 0, stream>>>(th_bf, pp_bf, f_f32);
  // 5) softmax rows
  nl_softmax_kernel<<<dim3((B * N) / 8), 256, 0, stream>>>(f_f32, at_bf);
  // 6) y = attn @ g (WMMA, 16x64 strips)
  nl_y_gemm_kernel<<<dim3(49, B), 256, 0, stream>>>(at_bf, gt_bf, y_bf);
  // 7) zero pooled accumulator
  nl_zero_kernel<<<dim3((B * C) / 256), 256, 0, stream>>>(pooled, B * C);
  // 8) wy GEMM + bias + BN + residual + column sums (WMMA, 16x64 strips)
  nl_out_gemm_kernel<<<dim3(98, B), 256, 0, stream>>>(
      y_bf, wo_bf, b_out, bn_gamma, bn_beta, bn_mean, bn_var, x, pooled);
  // 9) final FC
  nl_final_kernel<<<dim3(1), 64, 0, stream>>>(pooled, fc_w, fc_b, out);
}
